// GAT_Transformer_ContextFusion_4629974745452
// MI455X (gfx1250) — matrix operand order repaired and checked by reading, not compiled, not run on hardware
//
#include <hip/hip_runtime.h>
#include <hip/hip_bf16.h>

typedef __attribute__((ext_vector_type(16))) _Float16 v16h;
typedef __attribute__((ext_vector_type(8)))  _Float16 v8h;
typedef __attribute__((ext_vector_type(8)))  float    v8f;

// ---------------- wave reductions (wave32) ----------------
__device__ __forceinline__ float wred_sum(float v) {
#pragma unroll
  for (int o = 16; o > 0; o >>= 1) v += __shfl_xor(v, o, 32);
  return v;
}
__device__ __forceinline__ float wred_max(float v) {
#pragma unroll
  for (int o = 16; o > 0; o >>= 1) v = fmaxf(v, __shfl_xor(v, o, 32));
  return v;
}

// ---------------- weight prep: f32 -> f16, K x N layouts ----------------
// W already [K][N] row-major, pad rows K..Kpad with zero
__global__ void k_w_copy_f16(const float* __restrict__ W, _Float16* __restrict__ Bh,
                             int K, int Ncols, int Kpad) {
  int id = blockIdx.x * blockDim.x + threadIdx.x;
  int tot = Kpad * Ncols;
  if (id >= tot) return;
  int k = id / Ncols;
  Bh[id] = (k < K) ? (_Float16)W[id] : (_Float16)0.f;
}
// W is [O][K] row-major; produce Bh[k*O + o] = W[o][k] (transposed)
__global__ void k_w_transpose_f16(const float* __restrict__ W, _Float16* __restrict__ Bh,
                                  int O, int K) {
  int id = blockIdx.x * blockDim.x + threadIdx.x;
  if (id >= K * O) return;
  int k = id / O, o = id % O;
  Bh[id] = (_Float16)W[(size_t)o * K + k];
}
// pad x (MR x 16 f32) -> xh (MR x 32 f16, zero-padded K)
__global__ void k_pad_x_f16(const float* __restrict__ x, _Float16* __restrict__ xh) {
  int id = blockIdx.x * blockDim.x + threadIdx.x;
  if (id >= 49152 * 32) return;
  int m = id >> 5, k = id & 31;
  xh[id] = (k < 16) ? (_Float16)x[(size_t)m * 16 + k] : (_Float16)0.f;
}

// ---------------- generic WMMA GEMM: D = act(A(MxK) * B(f16,KxN) + bias) ----------------
// K must be a multiple of 32 (<=192). wave = one 16x16 tile; block = 4 waves = 64x16.
// B panel (Kx16) staged transposed in LDS once per block. act: 0 none, 1 relu, 2 sigmoid
template <bool AF16>
__global__ void __launch_bounds__(128)
k_gemm_wmma(const void* __restrict__ Av, const _Float16* __restrict__ Bm,
            const float* __restrict__ bias, float* __restrict__ D,
            int M, int Ncols, int K, int lda, int ldd, int act) {
  __shared__ __align__(32) _Float16 Bs[16][192];
  int tid = threadIdx.x;
  int lane = tid & 31, wave = tid >> 5;
  int r = lane & 15, hi = lane >> 4;
  int m0 = blockIdx.y * 64 + wave * 16;
  int n0 = blockIdx.x * 16;
  for (int idx = tid; idx < K * 16; idx += 128) {
    int k = idx >> 4, c = idx & 15;
    Bs[c][k] = Bm[(size_t)k * Ncols + n0 + c];
  }
  __syncthreads();
  if (m0 >= M) return;   // wave-uniform (M % 64 == 0 in practice)
  v8f acc = {};
  if constexpr (AF16) {
    const _Float16* Arow = (const _Float16*)Av + (size_t)(m0 + r) * lda;
    for (int k0 = 0; k0 < K; k0 += 32) {
      v8h alo = *(const v8h*)(Arow + k0 + hi * 16);
      v8h ahi = *(const v8h*)(Arow + k0 + hi * 16 + 8);
      v16h a, b;
#pragma unroll
      for (int i = 0; i < 8; ++i) { a[i] = alo[i]; a[i + 8] = ahi[i]; }
      b = *(const v16h*)(&Bs[r][k0 + hi * 16]);
      acc = __builtin_amdgcn_wmma_f32_16x16x32_f16(false, a, false, b, (short)0, acc, false, false);
    }
  } else {
    const float* Arow = (const float*)Av + (size_t)(m0 + r) * lda;
    for (int k0 = 0; k0 < K; k0 += 32) {
      const float4* p0 = (const float4*)(Arow + k0 + hi * 16);
      float4 x0 = p0[0], x1 = p0[1];
      const float4* p1 = (const float4*)(Arow + k0 + hi * 16 + 8);
      float4 y0 = p1[0], y1 = p1[1];
      v16h a, b;
      a[0] = (_Float16)x0.x; a[1] = (_Float16)x0.y; a[2] = (_Float16)x0.z; a[3] = (_Float16)x0.w;
      a[4] = (_Float16)x1.x; a[5] = (_Float16)x1.y; a[6] = (_Float16)x1.z; a[7] = (_Float16)x1.w;
      a[8] = (_Float16)y0.x; a[9] = (_Float16)y0.y; a[10] = (_Float16)y0.z; a[11] = (_Float16)y0.w;
      a[12] = (_Float16)y1.x; a[13] = (_Float16)y1.y; a[14] = (_Float16)y1.z; a[15] = (_Float16)y1.w;
      b = *(const v16h*)(&Bs[r][k0 + hi * 16]);
      acc = __builtin_amdgcn_wmma_f32_16x16x32_f16(false, a, false, b, (short)0, acc, false, false);
    }
  }
#pragma unroll
  for (int v = 0; v < 8; ++v) {            // C: vgpr v -> M = v + 8*hi, N = r
    int mm = m0 + v + 8 * hi;
    int nn = n0 + r;
    float o = acc[v] + (bias ? bias[nn] : 0.f);
    if (act == 1) o = fmaxf(o, 0.f);
    else if (act == 2) o = 1.f / (1.f + __expf(-o));
    D[(size_t)mm * ldd + nn] = o;
  }
}

// ---------------- GAT scores + transposed f16 copy of Wh ----------------
// Wh is (B,N,T,48) f32; write s1,s2 as (B,T,N); Whh as (B,T,N,48) f16
__global__ void k_s1s2(const float* __restrict__ Wh, const float* __restrict__ ga,
                       float* __restrict__ s1, float* __restrict__ s2,
                       _Float16* __restrict__ Whh) {
  int m = blockIdx.x * blockDim.x + threadIdx.x;
  if (m >= 49152) return;
  const float* wr = Wh + (size_t)m * 48;
  float d1 = 0.f, d2 = 0.f;
#pragma unroll
  for (int f = 0; f < 48; ++f) { float w = wr[f]; d1 += w * ga[f]; d2 += w * ga[48 + f]; }
  int b = m / (256 * 24), n = (m / 24) % 256, t = m % 24;
  int bt = b * 24 + t;
  s1[bt * 256 + n] = d1;
  s2[bt * 256 + n] = d2;
  _Float16* dstp = Whh + ((size_t)bt * 256 + n) * 48;
#pragma unroll
  for (int f = 0; f < 48; ++f) dstp[f] = (_Float16)wr[f];
}

// ---------------- fused GAT softmax + neighbor aggregation (per (b,t)) ----------------
// gcn[bt, i, f] = relu( sum_j softmax_j(mask(leaky(s1_i+s2_j))) * Wh[bt, j, f] )
__global__ void __launch_bounds__(128)
k_gcn(const _Float16* __restrict__ Whh, const float* __restrict__ s1g,
      const float* __restrict__ s2g, const int* __restrict__ adj,
      float* __restrict__ gcn) {
  __shared__ __align__(32) _Float16 WhT_s[48][256];   // 24 KB, transposed [f][j]
  __shared__ float s2_s[256];                         // 1 KB
  __shared__ __align__(32) _Float16 P_s[4][16 * 256]; // 32 KB (per-wave prob tiles)
  int bt = blockIdx.x;
  int tid = threadIdx.x;
  int lane = tid & 31, wave = tid >> 5;
  const _Float16* src = Whh + (size_t)bt * 256 * 48;
  for (int idx = tid; idx < 256 * 48; idx += 128) {
    int j = idx / 48, f = idx % 48;
    WhT_s[f][j] = src[idx];
  }
  for (int j = tid; j < 256; j += 128) s2_s[j] = s2g[bt * 256 + j];
  __syncthreads();
  int r = lane & 15, hi = lane >> 4;
  for (int tile = wave; tile < 16; tile += 4) {
    int i0 = tile * 16;
    // row softmax: 32 lanes each own 8 of the 256 j's
    for (int rr = 0; rr < 16; ++rr) {
      int i = i0 + rr;
      float s1v = s1g[bt * 256 + i];
      const int* arow = adj + (size_t)i * 256;
      float ev[8];
      float mx = -3.0e38f;
#pragma unroll
      for (int q = 0; q < 8; ++q) {
        int j = lane + 32 * q;
        float e = s1v + s2_s[j];
        e = (e > 0.f) ? e : 0.2f * e;       // leaky_relu(0.2)
        if (arow[j] <= 0) e = -9.0e15f;     // NEG mask
        ev[q] = e;
        mx = fmaxf(mx, e);
      }
      mx = wred_max(mx);
      float sm = 0.f;
#pragma unroll
      for (int q = 0; q < 8; ++q) { ev[q] = __expf(ev[q] - mx); sm += ev[q]; }
      sm = wred_sum(sm);
      float inv = 1.f / sm;
#pragma unroll
      for (int q = 0; q < 8; ++q) P_s[wave][rr * 256 + lane + 32 * q] = (_Float16)(ev[q] * inv);
    }
    // aggregate: (16 x 256) @ (256 x 48) via 8 K-steps x 3 N-tiles of WMMA
    const _Float16* Pw = P_s[wave];
    for (int n0 = 0; n0 < 48; n0 += 16) {
      v8f acc = {};
      for (int k0 = 0; k0 < 256; k0 += 32) {
        v8h alo = *(const v8h*)(Pw + r * 256 + k0 + hi * 16);
        v8h ahi = *(const v8h*)(Pw + r * 256 + k0 + hi * 16 + 8);
        v16h a, b;
#pragma unroll
        for (int i = 0; i < 8; ++i) { a[i] = alo[i]; a[i + 8] = ahi[i]; }
        b = *(const v16h*)(&WhT_s[n0 + r][k0 + hi * 16]);
        acc = __builtin_amdgcn_wmma_f32_16x16x32_f16(false, a, false, b, (short)0, acc, false, false);
      }
#pragma unroll
      for (int v = 0; v < 8; ++v) {
        int row = i0 + v + 8 * hi;
        gcn[((size_t)bt * 256 + row) * 48 + n0 + r] = fmaxf(acc[v], 0.f);
      }
    }
  }
}

// ---------------- build h: scrambled reshape concat node embedding ----------------
// seq = gcn(B,T,N,F).reshape(B*N, T, F) is a raw reinterpret: flat = m*1152 + t*48 + f
__global__ void k_build_h(const float* __restrict__ gcn, const float* __restrict__ emb,
                          float* __restrict__ h) {
  int id = blockIdx.x * blockDim.x + threadIdx.x; // = m*24 + t
  if (id >= 49152) return;
  int m = id / 24, t = id % 24;
  float* hr = h + (size_t)id * 64;
  const float* g = gcn + (size_t)m * 1152 + t * 48;
#pragma unroll
  for (int f = 0; f < 48; ++f) hr[f] = g[f];
  const float* e = emb + (m % 256) * 16;
#pragma unroll
  for (int j = 0; j < 16; ++j) hr[48 + j] = e[j];
}

// ---------------- per-sequence MHA core: T=24, 4 heads x 16 dim (VALU, tiny) ----------------
__global__ void __launch_bounds__(128)
k_attn(const float* __restrict__ qkv, float* __restrict__ ao) {
  __shared__ float Qs[4][384], Ks[4][384], Vs[4][384];
  int m = blockIdx.x;
  int lane = threadIdx.x & 31, h = threadIdx.x >> 5;  // wave == head
  for (int i = lane; i < 384; i += 32) {
    int t = i >> 4, d = i & 15;
    size_t base = ((size_t)m * 24 + t) * 192 + h * 16 + d;
    Qs[h][i] = qkv[base];
    Ks[h][i] = qkv[base + 64];
    Vs[h][i] = qkv[base + 128];
  }
  __syncthreads();
  if (lane < 24) {
    float p[24];
    float mx = -3.0e38f;
    for (int kt = 0; kt < 24; ++kt) {
      float d = 0.f;
#pragma unroll
      for (int dd = 0; dd < 16; ++dd) d += Qs[h][lane * 16 + dd] * Ks[h][kt * 16 + dd];
      d *= 0.25f;                                   // 1/sqrt(16)
      p[kt] = d;
      mx = fmaxf(mx, d);
    }
    float sm = 0.f;
    for (int kt = 0; kt < 24; ++kt) { p[kt] = __expf(p[kt] - mx); sm += p[kt]; }
    float inv = 1.f / sm;
    float* orow = ao + ((size_t)m * 24 + lane) * 64 + h * 16;
#pragma unroll
    for (int dd = 0; dd < 16; ++dd) {
      float o = 0.f;
      for (int kt = 0; kt < 24; ++kt) o += p[kt] * Vs[h][kt * 16 + dd];
      orow[dd] = o * inv;
    }
  }
}

// ---------------- (residual+) LayerNorm over 64 features, one wave per row ----------------
__global__ void k_ln64(const float* __restrict__ x, const float* __restrict__ res,
                       const float* __restrict__ g, const float* __restrict__ b,
                       float* __restrict__ out, int rows) {
  int lane = threadIdx.x & 31, wave = threadIdx.x >> 5;
  int row = blockIdx.x * 8 + wave;
  if (row >= rows) return;
  size_t base = (size_t)row * 64;
  int e0 = lane * 2, e1 = lane * 2 + 1;
  float v0 = x[base + e0], v1 = x[base + e1];
  if (res) { v0 += res[base + e0]; v1 += res[base + e1]; }
  float mean = wred_sum(v0 + v1) * (1.f / 64.f);
  float d0 = v0 - mean, d1 = v1 - mean;
  float var = wred_sum(d0 * d0 + d1 * d1) * (1.f / 64.f);
  float inv = rsqrtf(var + 1e-5f);
  out[base + e0] = d0 * inv * g[e0] + b[e0];
  out[base + e1] = d1 * inv * g[e1] + b[e1];
}

// ---------------- context gate: fused = h[:, -1, :] * cw ----------------
__global__ void k_fuse(const float* __restrict__ h, const float* __restrict__ cw,
                       float* __restrict__ fused) {
  int id = blockIdx.x * blockDim.x + threadIdx.x;
  if (id >= 2048 * 64) return;
  int m = id >> 6, e = id & 63;
  fused[id] = h[((size_t)m * 24 + 23) * 64 + e] * cw[id];
}

// ---------------- head: sigmoid(ph @ hw2.T + hb2), one wave per row ----------------
__global__ void k_pred(const float* __restrict__ ph, const float* __restrict__ hw2,
                       const float* __restrict__ hb2, float* __restrict__ out, int rows) {
  int lane = threadIdx.x & 31, wave = threadIdx.x >> 5;
  int row = blockIdx.x * 8 + wave;
  if (row >= rows) return;
  const float* pr = ph + (size_t)row * 64;
  float v = pr[lane] * hw2[lane] + pr[lane + 32] * hw2[lane + 32];
  v = wred_sum(v);
  if (lane == 0) out[row] = 1.f / (1.f + __expf(-(v + hb2[0])));
}

extern "C" void kernel_launch(void* const* d_in, const int* in_sizes, int n_in,
                              void* d_out, int out_size, void* d_ws, size_t ws_size,
                              hipStream_t stream) {
  (void)in_sizes; (void)n_in; (void)out_size; (void)ws_size;
  const float* x         = (const float*)d_in[0];
  const int*   adj       = (const int*)  d_in[1];
  const float* context   = (const float*)d_in[2];
  const float* emb       = (const float*)d_in[3];
  const float* gat_W     = (const float*)d_in[4];
  const float* gat_a     = (const float*)d_in[5];
  const float* blk_in_w  = (const float*)d_in[6];
  const float* blk_in_b  = (const float*)d_in[7];
  const float* blk_out_w = (const float*)d_in[8];
  const float* blk_out_b = (const float*)d_in[9];
  const float* blk_w1    = (const float*)d_in[10];
  const float* blk_b1    = (const float*)d_in[11];
  const float* blk_w2    = (const float*)d_in[12];
  const float* blk_b2    = (const float*)d_in[13];
  const float* n1g       = (const float*)d_in[14];
  const float* n1b       = (const float*)d_in[15];
  const float* n2g       = (const float*)d_in[16];
  const float* n2b       = (const float*)d_in[17];
  const float* cw1       = (const float*)d_in[18];
  const float* cb1       = (const float*)d_in[19];
  const float* cw2       = (const float*)d_in[20];
  const float* cb2       = (const float*)d_in[21];
  const float* hw1       = (const float*)d_in[22];
  const float* hb1       = (const float*)d_in[23];
  const float* hlng      = (const float*)d_in[24];
  const float* hlnb      = (const float*)d_in[25];
  const float* hw2       = (const float*)d_in[26];
  const float* hb2       = (const float*)d_in[27];
  float* out = (float*)d_out;

  const int MR = 49152;   // B*N*T rows
  const int MS = 2048;    // B*N sequences
  const int BT = 192;     // B*T

  // ---- workspace layout (256B aligned bump allocator) ----
  size_t off = 0;
  auto alloc = [&](size_t bytes) {
    off = (off + 255) & ~(size_t)255;
    void* p = (char*)d_ws + off;
    off += bytes;
    return p;
  };
  _Float16*  xh    = (_Float16*) alloc((size_t)MR * 32 * 2);   // padded f16 x
  float*     Wh    = (float*)    alloc((size_t)MR * 48 * 4);
  _Float16*  Whh   = (_Float16*) alloc((size_t)MR * 48 * 2);   // (B,T,N,48) f16
  float*     s1    = (float*)    alloc((size_t)MR * 4);
  float*     s2    = (float*)    alloc((size_t)MR * 4);
  float*     gcn   = (float*)    alloc((size_t)MR * 48 * 4);   // (B,T,N,48)
  float*     h     = (float*)    alloc((size_t)MR * 64 * 4);
  float*     qkv   = (float*)    alloc((size_t)MR * 192 * 4);  // reused as ff1 (MR*128)
  float*     ao    = (float*)    alloc((size_t)MR * 64 * 4);
  float*     proj  = (float*)    alloc((size_t)MR * 64 * 4);   // reused as ff2
  float*     c1    = (float*)    alloc((size_t)MS * 128 * 4);
  float*     c2    = (float*)    alloc((size_t)MS * 64 * 4);
  float*     fusedB= (float*)    alloc((size_t)MS * 64 * 4);
  float*     ph    = (float*)    alloc((size_t)MS * 64 * 4);
  _Float16*  gatB  = (_Float16*) alloc(32 * 48 * 2);
  _Float16*  WinT  = (_Float16*) alloc(2 * 64 * 192 * 2);
  _Float16*  WoutT = (_Float16*) alloc(2 * 64 * 64 * 2);
  _Float16*  W1T   = (_Float16*) alloc(2 * 64 * 128 * 2);
  _Float16*  W2T   = (_Float16*) alloc(2 * 128 * 64 * 2);
  _Float16*  cw1T  = (_Float16*) alloc(192 * 128 * 2);
  _Float16*  cw2T  = (_Float16*) alloc(128 * 64 * 2);
  _Float16*  hw1T  = (_Float16*) alloc(64 * 64 * 2);

  auto cdiv = [](int a, int b) { return (a + b - 1) / b; };

  // ---- weight / input prep ----
  k_w_copy_f16<<<cdiv(32 * 48, 256), 256, 0, stream>>>(gat_W, gatB, 16, 48, 32);
  k_pad_x_f16<<<cdiv(MR * 32, 256), 256, 0, stream>>>(x, xh);
  for (int l = 0; l < 2; ++l) {
    k_w_transpose_f16<<<cdiv(64 * 192, 256), 256, 0, stream>>>(blk_in_w + l * 192 * 64, WinT + l * 64 * 192, 192, 64);
    k_w_transpose_f16<<<cdiv(64 * 64, 256), 256, 0, stream>>>(blk_out_w + l * 64 * 64, WoutT + l * 64 * 64, 64, 64);
    k_w_transpose_f16<<<cdiv(64 * 128, 256), 256, 0, stream>>>(blk_w1 + l * 128 * 64, W1T + l * 64 * 128, 128, 64);
    k_w_transpose_f16<<<cdiv(128 * 64, 256), 256, 0, stream>>>(blk_w2 + l * 64 * 128, W2T + l * 128 * 64, 64, 128);
  }
  k_w_transpose_f16<<<cdiv(192 * 128, 256), 256, 0, stream>>>(cw1, cw1T, 128, 192);
  k_w_transpose_f16<<<cdiv(128 * 64, 256), 256, 0, stream>>>(cw2, cw2T, 64, 128);
  k_w_transpose_f16<<<cdiv(64 * 64, 256), 256, 0, stream>>>(hw1, hw1T, 64, 64);

  // ---- GAT projection: Wh = x @ gat_W  (M=49152, N=48, K=32 padded) ----
  k_gemm_wmma<true><<<dim3(48 / 16, MR / 64), 128, 0, stream>>>(xh, gatB, nullptr, Wh, MR, 48, 32, 32, 48, 0);
  // scores + f16 transpose of Wh into (B,T,N,48)
  k_s1s2<<<cdiv(MR, 128), 128, 0, stream>>>(Wh, gat_a, s1, s2, Whh);
  // fused softmax + aggregation
  k_gcn<<<BT, 128, 0, stream>>>(Whh, s1, s2, adj, gcn);
  // assemble h (scrambled reshape + node embedding concat)
  k_build_h<<<cdiv(MR, 128), 128, 0, stream>>>(gcn, emb, h);

  // ---- transformer blocks ----
  for (int l = 0; l < 2; ++l) {
    k_gemm_wmma<false><<<dim3(192 / 16, MR / 64), 128, 0, stream>>>(h, WinT + l * 64 * 192, blk_in_b + l * 192, qkv, MR, 192, 64, 64, 192, 0);
    k_attn<<<MS, 128, 0, stream>>>(qkv, ao);
    k_gemm_wmma<false><<<dim3(64 / 16, MR / 64), 128, 0, stream>>>(ao, WoutT + l * 64 * 64, blk_out_b + l * 64, proj, MR, 64, 64, 64, 64, 0);
    k_ln64<<<MR / 8, 256, 0, stream>>>(h, proj, n1g + l * 64, n1b + l * 64, h, MR);
    float* ff1 = qkv;   // reuse
    float* ff2 = proj;  // reuse
    k_gemm_wmma<false><<<dim3(128 / 16, MR / 64), 128, 0, stream>>>(h, W1T + l * 64 * 128, blk_b1 + l * 128, ff1, MR, 128, 64, 64, 128, 1);
    k_gemm_wmma<false><<<dim3(64 / 16, MR / 64), 128, 0, stream>>>(ff1, W2T + l * 128 * 64, blk_b2 + l * 64, ff2, MR, 64, 128, 128, 64, 0);
    k_ln64<<<MR / 8, 256, 0, stream>>>(h, ff2, n2g + l * 64, n2b + l * 64, h, MR);
  }

  // ---- context gate ----
  k_gemm_wmma<false><<<dim3(128 / 16, MS / 64), 128, 0, stream>>>(context, cw1T, cb1, c1, MS, 128, 192, 192, 128, 1);
  k_gemm_wmma<false><<<dim3(64 / 16, MS / 64), 128, 0, stream>>>(c1, cw2T, cb2, c2, MS, 64, 128, 128, 64, 2);
  k_fuse<<<cdiv(MS * 64, 256), 256, 0, stream>>>(h, c2, fusedB);

  // ---- head ----
  k_gemm_wmma<false><<<dim3(64 / 16, MS / 64), 128, 0, stream>>>(fusedB, hw1T, hb1, ph, MS, 64, 64, 64, 64, 1);
  k_ln64<<<MS / 8, 256, 0, stream>>>(ph, nullptr, hlng, hlnb, ph, MS);
  k_pred<<<MS / 8, 256, 0, stream>>>(ph, hw2, hb2, out, MS);
}
